// RGCNLayer_6906307412500
// MI455X (gfx1250) — matrix operand order repaired and checked by reading, not compiled, not run on hardware
//
#include <hip/hip_runtime.h>
#include <hip/hip_bf16.h>

typedef __attribute__((ext_vector_type(16))) _Float16 v16h;
typedef __attribute__((ext_vector_type(8)))  float    v8f;

#define IN_FEAT  128
#define OUT_FEAT 128
#define NUM_RELS 8

// ---------------------------------------------------------------------------
// f32 -> f16 conversion of node features
// ---------------------------------------------------------------------------
__global__ void cvt_feat_kernel(const float* __restrict__ feat,
                                _Float16* __restrict__ featH, int n) {
    int i = blockIdx.x * blockDim.x + threadIdx.x;
    if (i < n) featH[i] = (_Float16)feat[i];
}

// ---------------------------------------------------------------------------
// f32 -> f16 + transpose of the 9 weight matrices (8 relations + self-loop).
// wT[r][o][k] = W[r][k][o]  so that B fragments load contiguously along K.
// ---------------------------------------------------------------------------
__global__ void cvt_weight_kernel(const float* __restrict__ w,
                                  const float* __restrict__ lw,
                                  _Float16* __restrict__ wT) {
    int i = blockIdx.x * blockDim.x + threadIdx.x;
    if (i >= (NUM_RELS + 1) * IN_FEAT * OUT_FEAT) return;
    int r   = i >> 14;            // /16384
    int rem = i & 16383;
    int o   = rem >> 7;           // out index (row of wT)
    int k   = rem & 127;          // in index (contiguous in wT)
    float v = (r < NUM_RELS) ? w[(r << 14) + (k << 7) + o]
                             : lw[(k << 7) + o];
    wT[i] = (_Float16)v;
}

// ---------------------------------------------------------------------------
// Grouped GEMM: for r in 0..8, out_r = featH @ wT[r]^T  (16x16x32 f16 WMMA).
// One wave per (relation, 16-row node tile), computing a 16x128 strip.
// Relations 0..7 -> all_h workspace; relation 8 (self-loop) -> d_out directly.
// ---------------------------------------------------------------------------
__global__ void __launch_bounds__(128)
rgcn_gemm_wmma(const _Float16* __restrict__ featH,
               const _Float16* __restrict__ wT,
               float* __restrict__ all_h,
               float* __restrict__ out,
               int n_nodes, int m_tiles) {
    const int lane = threadIdx.x;                       // 0..31 (wave32)
    const int wave = blockIdx.x * blockDim.y + threadIdx.y;
    const int total = (NUM_RELS + 1) * m_tiles;
    if (wave >= total) return;

    const int r      = wave / m_tiles;
    const int mt     = wave - r * m_tiles;
    const int m_base = mt * 16;

    // A fragment addressing: 16-bit A 16x32 layout (wave32):
    // lane L: row = L%16; halves 0..7 = K(koff..koff+7), 8..15 = K(koff+16..+23)
    // where koff = (L/16)*8.
    int a_row = m_base + (lane & 15);
    if (a_row >= n_nodes) a_row = n_nodes - 1;          // clamp (safe garbage)
    const _Float16* a_ptr = featH + (size_t)a_row * IN_FEAT + ((lane >> 4) << 3);

    // B fragment addressing: lane L holds col n = L%16; halves 0..7 = K(koff..koff+7),
    // 8..15 = K(koff+16..+23), koff = (L/16)*8. wT row o is contiguous in K.
    const _Float16* b_base = wT + (size_t)r * IN_FEAT * OUT_FEAT
                                + (size_t)(lane & 15) * IN_FEAT
                                + ((lane >> 4) << 3);

    v8f acc[8] = {};

#pragma unroll
    for (int k0 = 0; k0 < IN_FEAT; k0 += 32) {
        v16h a;
        *((float4*)&a)       = *(const float4*)(a_ptr + k0);
        *(((float4*)&a) + 1) = *(const float4*)(a_ptr + k0 + 16);
#pragma unroll
        for (int nt = 0; nt < 8; ++nt) {
            const _Float16* bp = b_base + (size_t)nt * 16 * IN_FEAT + k0;
            v16h b;
            *((float4*)&b)       = *(const float4*)(bp);
            *(((float4*)&b) + 1) = *(const float4*)(bp + 16);
            acc[nt] = __builtin_amdgcn_wmma_f32_16x16x32_f16(
                false, a, false, b, (short)0, acc[nt], false, false);
        }
    }

    // C/D layout: VGPR j -> M = j (lanes 0-15) or M = 8+j (lanes 16-31),
    // N = lane%16.
    float* dstmat = (r < NUM_RELS) ? (all_h + (size_t)r * n_nodes * OUT_FEAT)
                                   : out;
    const int ncol  = lane & 15;
    const int mrow0 = m_base + ((lane >> 4) << 3);
#pragma unroll
    for (int nt = 0; nt < 8; ++nt) {
#pragma unroll
        for (int j = 0; j < 8; ++j) {
            int m = mrow0 + j;
            if (m < n_nodes)
                dstmat[(size_t)m * OUT_FEAT + nt * 16 + ncol] = acc[nt][j];
        }
    }
}

// ---------------------------------------------------------------------------
// Edge pass: one wave per edge. Gather 128 floats of all_h[etype, src],
// scale by norm, atomic-add into out[dst]. float4 load per lane (coalesced),
// 4 non-returning global_atomic_add_f32 per lane.
// ---------------------------------------------------------------------------
__global__ void __launch_bounds__(256)
edge_scatter(const float* __restrict__ all_h,
             const float* __restrict__ norm,
             const int* __restrict__ src,
             const int* __restrict__ dst,
             const int* __restrict__ etypes,
             float* __restrict__ out,
             int n_edges, int n_nodes) {
    const int lane = threadIdx.x;                       // 0..31
    const int e    = blockIdx.x * blockDim.y + threadIdx.y;
    if (e >= n_edges) return;

    const int   r = etypes[e];
    const int   s = src[e];
    const int   d = dst[e];
    const float w = norm[e];

    const float4* row =
        (const float4*)(all_h + ((size_t)r * n_nodes + s) * OUT_FEAT);
    float4 v = row[lane];

    float* o = out + (size_t)d * OUT_FEAT + lane * 4;
    atomicAdd(o + 0, v.x * w);
    atomicAdd(o + 1, v.y * w);
    atomicAdd(o + 2, v.z * w);
    atomicAdd(o + 3, v.w * w);
}

// ---------------------------------------------------------------------------
extern "C" void kernel_launch(void* const* d_in, const int* in_sizes, int n_in,
                              void* d_out, int out_size, void* d_ws, size_t ws_size,
                              hipStream_t stream) {
    const float* feat        = (const float*)d_in[0];
    const float* weight      = (const float*)d_in[1];
    const float* loop_weight = (const float*)d_in[2];
    const float* norm        = (const float*)d_in[3];
    const int*   src         = (const int*)d_in[4];
    const int*   dst         = (const int*)d_in[5];
    const int*   etypes      = (const int*)d_in[6];
    float*       out         = (float*)d_out;

    const int n_nodes = in_sizes[0] / IN_FEAT;
    const int n_edges = in_sizes[4];

    // Workspace layout: all_h (f32) | featH (f16) | wT (f16), 256B aligned.
    char*  ws          = (char*)d_ws;
    float* all_h       = (float*)ws;
    size_t all_h_bytes = (size_t)NUM_RELS * n_nodes * OUT_FEAT * sizeof(float);
    _Float16* featH    = (_Float16*)(ws + ((all_h_bytes + 255) & ~(size_t)255));
    size_t featH_bytes = (size_t)n_nodes * IN_FEAT * sizeof(_Float16);
    _Float16* wT       = (_Float16*)((char*)featH +
                                     ((featH_bytes + 255) & ~(size_t)255));

    // 1) precision conversion + weight transpose
    const int nfe = n_nodes * IN_FEAT;
    cvt_feat_kernel<<<(nfe + 255) / 256, 256, 0, stream>>>(feat, featH, nfe);
    const int nwe = (NUM_RELS + 1) * IN_FEAT * OUT_FEAT;
    cvt_weight_kernel<<<(nwe + 255) / 256, 256, 0, stream>>>(weight, loop_weight, wT);

    // 2) grouped WMMA GEMM (rel 8 = self-loop writes d_out, initializing it)
    const int m_tiles     = (n_nodes + 15) / 16;
    const int total_waves = (NUM_RELS + 1) * m_tiles;
    dim3 gblock(32, 4);   // 4 waves per block
    rgcn_gemm_wmma<<<(total_waves + 3) / 4, gblock, 0, stream>>>(
        featH, wT, all_h, out, n_nodes, m_tiles);

    // 3) edge gather + atomic scatter
    dim3 eblock(32, 8);   // 8 edges per block
    edge_scatter<<<(n_edges + 7) / 8, eblock, 0, stream>>>(
        all_h, norm, src, dst, etypes, out, n_edges, n_nodes);
}
